// MultiHeadAttention_73203422593073
// MI455X (gfx1250) — matrix operand order repaired and checked by reading, not compiled, not run on hardware
//
#include <hip/hip_runtime.h>
#include <hip/hip_bf16.h>

#define EMB   256
#define HEADS 8
#define HD    32
#define BATCH 2
#define SEQ   4096
#define GS    (BATCH * SEQ) /* 8192 */

typedef _Float16 v4h  __attribute__((ext_vector_type(4)));
typedef _Float16 v8h  __attribute__((ext_vector_type(8)));
typedef _Float16 v16h __attribute__((ext_vector_type(16)));
typedef float    v8f  __attribute__((ext_vector_type(8)));
typedef unsigned int u32x4 __attribute__((ext_vector_type(4)));
typedef int      i32x4 __attribute__((ext_vector_type(4)));
typedef int      i32x8 __attribute__((ext_vector_type(8)));

union H16 { v16h v; v8h h[2]; };

#define EXP2 __builtin_amdgcn_exp2f

__device__ __forceinline__ v8f wmma16(v16h a, v16h b, v8f c) {
  // D = A(16x32 f16) * B(32x16 f16) + C(16x16 f32)
  return __builtin_amdgcn_wmma_f32_16x16x32_f16(false, a, false, b, (short)0, c,
                                                false, false);
}

// A fragment (16x32 f16) from row-major source. ISA 7.12.2: lane r holds row r;
// halves h[0]: K = g*8 .. g*8+7, h[1]: K = 16+g*8 .. 16+g*8+7 (g = lane>>4).
__device__ __forceinline__ v16h load_a(const _Float16* row0, int lda, int r, int g) {
  H16 u;
  const _Float16* p = row0 + (size_t)r * lda + g * 8;
  u.h[0] = *(const v8h*)(p);
  u.h[1] = *(const v8h*)(p + 16);
  return u.v;
}

// B fragment (32x16 f16) where B(k,n) = src[n*ldb + k]. Lane holds column n;
// halves h[0]: K = g*8 .. g*8+7, h[1]: K = 16+g*8 .. 16+g*8+7 (matches load_a).
__device__ __forceinline__ v16h load_b(const _Float16* base, int ldb, int n, int g) {
  H16 u;
  const _Float16* p = base + (size_t)n * ldb + g * 8;
  u.h[0] = *(const v8h*)(p);
  u.h[1] = *(const v8h*)(p + 16);
  return u.v;
}

// ---- TDM: load a 32-row x 64B tile (one K block) into LDS, padded to 80B rows.
// D# per ISA 8.3/8.4: data_size=4B; tile_dim0=16 dw; tile_dim1=32 rows;
// tensor_dim0_stride=16 dw; pad_enable, 4 dwords every 16 dwords (64B->80B rows).
__device__ __forceinline__ void tdm_load_k(unsigned lds_off,
                                           unsigned long long gaddr) {
  u32x4 g0;
  g0[0] = 1u;                                   // count = 1 (valid descriptor)
  g0[1] = lds_off;                              // lds_addr (bytes)
  g0[2] = (unsigned)gaddr;                      // global_addr[31:0]
  g0[3] = (unsigned)(gaddr >> 32) | (2u << 30); // global_addr[56:32] | type=2
  i32x8 g1;
  g1[0] = (2 << 16) | (1 << 20) | (3 << 22) | (3 << 25); // dsz=4B,pad 4dw/16dw
  g1[1] = (16 << 16);        // tensor_dim0 = 16 dw  (bits 79:48, low half here)
  g1[2] = (int)(4096u << 16);// tensor_dim1 = 4096 rows (bits 111:80, low half)
  g1[3] = (16 << 16);        // tensor_dim1 hi = 0 ; tile_dim0 = 16 dw
  g1[4] = 32;                // tile_dim1 = 32 rows ; tile_dim2 = 0
  g1[5] = 16;                // tensor_dim0_stride = 16 dw (64B K rows)
  g1[6] = 0;
  g1[7] = 0;
  i32x4 z4 = {};
#if defined(__clang_major__) && __clang_major__ >= 23
  i32x8 z8 = {};
  __builtin_amdgcn_tensor_load_to_lds(g0, g1, z4, z4, z8, 0);
#else
  __builtin_amdgcn_tensor_load_to_lds(g0, g1, z4, z4, 0);
#endif
}

// ---------------- f32 -> f16 conversion of the query activations ------------
__global__ void k_cvt_query(const float* __restrict__ q, _Float16* __restrict__ qh, int n) {
  int i = (blockIdx.x * blockDim.x + threadIdx.x) * 4;
  if (i + 3 < n) {
    float4 v = *(const float4*)(q + i);
    v4h o;
    o[0] = (_Float16)v.x; o[1] = (_Float16)v.y;
    o[2] = (_Float16)v.z; o[3] = (_Float16)v.w;
    *(v4h*)(qh + i) = o;
  }
}

// Weights: transpose to [h][d][e] f16 so B-fragment loads are contiguous.
// Q weights pre-scaled by (1/sqrt(HD)) * log2(e) so softmax uses pure exp2.
__global__ void k_cvt_weights(const float* __restrict__ wq, const float* __restrict__ wk,
                              const float* __restrict__ wv, const float* __restrict__ wo,
                              _Float16* __restrict__ WtQ, _Float16* __restrict__ WtK,
                              _Float16* __restrict__ WtV, _Float16* __restrict__ WoT) {
  int i = blockIdx.x * blockDim.x + threadIdx.x;
  if (i >= HEADS * EMB * HD) return;
  int h = i / (EMB * HD), e = (i / HD) % EMB, d = i % HD;
  const float QS = 0.17677669529663687f * 1.4426950408889634f;
  size_t t = ((size_t)h * HD + d) * EMB + e;
  WtQ[t] = (_Float16)(wq[i] * QS);
  WtK[t] = (_Float16)(wk[i]);
  WtV[t] = (_Float16)(wv[i]);
  // wo: [h][d][o] flat = hd*EMB + o  ->  WoT[o][hd]
  int hd = i / EMB, o = i % EMB;
  WoT[(size_t)o * (HEADS * HD) + hd] = (_Float16)(wo[(size_t)hd * EMB + o]);
}

// ---------------- QKV projection: [8192x256] x [256x32] per head ------------
__global__ void __launch_bounds__(128) k_qkv(const _Float16* __restrict__ qh,
                      const _Float16* __restrict__ WtQ, const _Float16* __restrict__ WtK,
                      const _Float16* __restrict__ WtV,
                      _Float16* __restrict__ Qh, _Float16* __restrict__ Kh,
                      _Float16* __restrict__ Vt) {
  int w = blockIdx.x * 4 + (threadIdx.x >> 5);
  int lane = threadIdx.x & 31, r = lane & 15, g = lane >> 4;
  int stile = w & 511;
  int h     = (w >> 9) & 7;
  int mat   = w >> 12;
  const _Float16* Wt = (mat == 0) ? WtQ : (mat == 1) ? WtK : WtV;
  const _Float16* arow = qh + (size_t)stile * 16 * EMB;
  const _Float16* wb   = Wt + (size_t)h * HD * EMB;
  v8f c0 = {}, c1 = {};
  #pragma unroll
  for (int kk = 0; kk < EMB; kk += 32) {
    v16h a  = load_a(arow + kk, EMB, r, g);
    v16h b0 = load_b(wb + kk, EMB, r, g);
    v16h b1 = load_b(wb + kk, EMB, 16 + r, g);
    c0 = wmma16(a, b0, c0);
    c1 = wmma16(a, b1, c1);
  }
  int srow0 = stile * 16;
  if (mat < 2) {
    _Float16* dst = (mat == 0) ? Qh : Kh;
    #pragma unroll
    for (int v = 0; v < 8; ++v) {
      size_t row = (size_t)h * GS + srow0 + v + 8 * g;
      dst[row * HD + r]      = (_Float16)c0[v];
      dst[row * HD + 16 + r] = (_Float16)c1[v];
    }
  } else {
    v8h o0, o1;
    #pragma unroll
    for (int v = 0; v < 8; ++v) { o0[v] = (_Float16)c0[v]; o1[v] = (_Float16)c1[v]; }
    size_t base = (size_t)h * HD * GS;
    *(v8h*)(Vt + base + (size_t)r * GS + srow0 + 8 * g)        = o0;
    *(v8h*)(Vt + base + (size_t)(16 + r) * GS + srow0 + 8 * g) = o1;
  }
}

// ---------------- fused flash attention with materialized probabilities -----
// One wave per workgroup: all TDM descriptor fields are wave-uniform (blockIdx
// only), so tensor_load_to_lds operands stay scalar. K tiles double-buffered
// in LDS via TDM; V fragments via global b128 (overlapping memory paths).
__global__ void __launch_bounds__(32) k_attn(const _Float16* __restrict__ Qh,
                      const _Float16* __restrict__ Kh, const _Float16* __restrict__ Vt,
                      float* __restrict__ attn, _Float16* __restrict__ heads) {
  __shared__ _Float16 kbuf[2][32 * 40]; // 32 K rows, 40-half (80B) padded stride
  __shared__ _Float16 pshm[16 * 40];    // P transpose tile
  int w = blockIdx.x;
  int lane = threadIdx.x & 31, r = lane & 15, g = lane >> 4;
  int qtile = w & 255;
  int b = (w >> 8) & 1;
  int h = w >> 9;
  const _Float16* QB = Qh + ((size_t)h * GS + b * SEQ + qtile * 16) * HD;
  const _Float16* VB = Vt + (size_t)h * HD * GS + (size_t)b * SEQ;
  unsigned long long gK =
      (unsigned long long)(size_t)(const void*)(Kh + ((size_t)h * GS + b * SEQ) * HD);
  unsigned ldsK[2] = { (unsigned)(size_t)(void*)&kbuf[0][0],
                       (unsigned)(size_t)(void*)&kbuf[1][0] };
  v16h aq = load_a(QB, HD, r, g);       // full head_dim = one A fragment, reused
  v8f zero = {};
  float m[8], s[8];
  #pragma unroll
  for (int v = 0; v < 8; ++v) { m[v] = -3.0e38f; s[v] = 0.f; }

  // ---- pass 1: online row max / sum (base 2; log2e folded into Q) ----
  tdm_load_k(ldsK[0], gK);
  int cur = 0;
  for (int k0 = 0; k0 < SEQ; k0 += 32) {
    if (k0 + 32 < SEQ) {
      tdm_load_k(ldsK[cur ^ 1], gK + (unsigned long long)(k0 + 32) * (HD * 2));
      __builtin_amdgcn_s_wait_tensorcnt(1);
    } else {
      __builtin_amdgcn_s_wait_tensorcnt(0);
    }
    const _Float16* kc = &kbuf[cur][0];
    v16h bk0 = load_b(kc, 40, r, g);
    v16h bk1 = load_b(kc, 40, 16 + r, g);
    v8f u0 = wmma16(aq, bk0, zero);
    v8f u1 = wmma16(aq, bk1, zero);
    #pragma unroll
    for (int v = 0; v < 8; ++v) {       // merged 2-value online update: 3 exp2
      float x0 = u0[v], x1 = u1[v];
      float mn = fmaxf(m[v], fmaxf(x0, x1));
      s[v] = s[v] * EXP2(m[v] - mn) + EXP2(x0 - mn) + EXP2(x1 - mn);
      m[v] = mn;
    }
    cur ^= 1;
  }
  // exact online-softmax merge across the 16 lanes owning each row
  float inv[8];
  #pragma unroll
  for (int v = 0; v < 8; ++v) {
    #pragma unroll
    for (int msk = 1; msk < 16; msk <<= 1) {
      float mo = __shfl_xor(m[v], msk, 16);
      float so = __shfl_xor(s[v], msk, 16);
      float mn = fmaxf(m[v], mo);
      s[v] = s[v] * EXP2(m[v] - mn) + so * EXP2(mo - mn);
      m[v] = mn;
    }
    inv[v] = 1.0f / s[v];
  }

  // ---- pass 2: recompute scores, write attention, fuse P*V ----
  v8f acc0 = {}, acc1 = {};
  float* A = attn + (((size_t)(h * BATCH + b) * SEQ) + qtile * 16) * SEQ;
  tdm_load_k(ldsK[0], gK);
  cur = 0;
  for (int k0 = 0; k0 < SEQ; k0 += 32) {
    if (k0 + 32 < SEQ) {
      tdm_load_k(ldsK[cur ^ 1], gK + (unsigned long long)(k0 + 32) * (HD * 2));
      __builtin_amdgcn_s_wait_tensorcnt(1);
    } else {
      __builtin_amdgcn_s_wait_tensorcnt(0);
    }
    const _Float16* kc = &kbuf[cur][0];
    #pragma unroll
    for (int t = 0; t < 2; ++t) {
      v16h bk = load_b(kc, 40, t * 16 + r, g);
      v8f u = wmma16(aq, bk, zero);
      #pragma unroll
      for (int v = 0; v < 8; ++v) {
        float p = EXP2(u[v] - m[v]) * inv[v];
        A[(size_t)(v + 8 * g) * SEQ + k0 + t * 16 + r] = p;  // coalesced 64B rows
        pshm[(v + 8 * g) * 40 + t * 16 + r] = (_Float16)p;   // LDS transpose stage
      }
    }
    v16h ap  = load_a(pshm, 40, r, g);                       // ds_load_b128 x2
    v16h bv0 = load_b(VB + k0, GS, r, g);
    v16h bv1 = load_b(VB + k0, GS, 16 + r, g);
    acc0 = wmma16(ap, bv0, acc0);
    acc1 = wmma16(ap, bv1, acc1);
    cur ^= 1;
  }
  #pragma unroll
  for (int v = 0; v < 8; ++v) {
    size_t row = (size_t)b * SEQ + qtile * 16 + v + 8 * g;
    heads[row * EMB + h * HD + r]      = (_Float16)acc0[v];
    heads[row * EMB + h * HD + 16 + r] = (_Float16)acc1[v];
  }
}

// ---------------- output projection: [8192x256] x [256x256] -----------------
__global__ void __launch_bounds__(128) k_out(const _Float16* __restrict__ heads,
                     const _Float16* __restrict__ WoT, float* __restrict__ out) {
  int w = blockIdx.x * 4 + (threadIdx.x >> 5);
  int lane = threadIdx.x & 31, r = lane & 15, g = lane >> 4;
  int stile = w & 511;
  int np    = w >> 9;
  const _Float16* arow = heads + (size_t)stile * 16 * EMB;
  v8f c0 = {}, c1 = {};
  #pragma unroll
  for (int kk = 0; kk < EMB; kk += 32) {
    v16h a  = load_a(arow + kk, EMB, r, g);
    v16h b0 = load_b(WoT + kk, EMB, np * 32 + r, g);
    v16h b1 = load_b(WoT + kk, EMB, np * 32 + 16 + r, g);
    c0 = wmma16(a, b0, c0);
    c1 = wmma16(a, b1, c1);
  }
  #pragma unroll
  for (int v = 0; v < 8; ++v) {
    size_t row = (size_t)stile * 16 + v + 8 * g;
    out[row * EMB + np * 32 + r]      = c0[v];
    out[row * EMB + np * 32 + 16 + r] = c1[v];
  }
}

extern "C" void kernel_launch(void* const* d_in, const int* in_sizes, int n_in,
                              void* d_out, int out_size, void* d_ws, size_t ws_size,
                              hipStream_t stream) {
  (void)in_sizes; (void)n_in; (void)out_size; (void)ws_size;
  const float* query = (const float*)d_in[0];
  const float* wq = (const float*)d_in[1];
  const float* wk = (const float*)d_in[2];
  const float* wv = (const float*)d_in[3];
  const float* wo = (const float*)d_in[4];
  float* out  = (float*)d_out;
  float* attn = out + (size_t)BATCH * SEQ * EMB;

  char* p = (char*)d_ws;
  auto alloc = [&](size_t bytes) {
    char* q = p; p += (bytes + 255) & ~(size_t)255; return q;
  };
  _Float16* qh    = (_Float16*)alloc((size_t)GS * EMB * 2);
  _Float16* WtQ   = (_Float16*)alloc((size_t)HEADS * HD * EMB * 2);
  _Float16* WtK   = (_Float16*)alloc((size_t)HEADS * HD * EMB * 2);
  _Float16* WtV   = (_Float16*)alloc((size_t)HEADS * HD * EMB * 2);
  _Float16* WoT   = (_Float16*)alloc((size_t)EMB * HEADS * HD * 2);
  _Float16* Qh    = (_Float16*)alloc((size_t)HEADS * GS * HD * 2);
  _Float16* Kh    = (_Float16*)alloc((size_t)HEADS * GS * HD * 2);
  _Float16* Vt    = (_Float16*)alloc((size_t)HEADS * HD * GS * 2);
  _Float16* headsB= (_Float16*)alloc((size_t)GS * EMB * 2);

  k_cvt_query<<<(GS * EMB / 4 + 255) / 256, 256, 0, stream>>>(query, qh, GS * EMB);
  k_cvt_weights<<<(HEADS * EMB * HD + 255) / 256, 256, 0, stream>>>(
      wq, wk, wv, wo, WtQ, WtK, WtV, WoT);
  k_qkv<<<(3 * HEADS * 512) / 4, 128, 0, stream>>>(qh, WtQ, WtK, WtV, Qh, Kh, Vt);
  k_attn<<<HEADS * BATCH * (SEQ / 16), 32, 0, stream>>>(Qh, Kh, Vt, attn, headsB);
  k_out<<<(512 * 8) / 4, 128, 0, stream>>>(headsB, WoT, out);
}
